// RTGMFFModel_33767032882069
// MI455X (gfx1250) — matrix operand order repaired and checked by reading, not compiled, not run on hardware
//
#include <hip/hip_runtime.h>
#include <math.h>
#include <stdint.h>

// ---------------------------------------------------------------------------
// Mamba-style SSM layer for MI455X (gfx1250), wave32 + WMMA bf16 + TDM.
// GEMMs: weight strip (32 x K) staged into LDS once per block via the Tensor
// Data Mover (tensor_load_to_lds, TENSORcnt), activations streamed from
// global with prefetch; B fragments served from LDS (ds_load_b128).
// ---------------------------------------------------------------------------

typedef __bf16 bf16;
typedef __attribute__((ext_vector_type(16))) __bf16 v16bf;
typedef __attribute__((ext_vector_type(8)))  __bf16 v8bf;
typedef __attribute__((ext_vector_type(8)))  float  v8f;
typedef __attribute__((ext_vector_type(4)))  uint32_t v4u;
typedef __attribute__((ext_vector_type(8)))  int      v8i;
typedef __attribute__((ext_vector_type(4)))  int      v4i;

#define BATCH   4
#define SEQ     2048
#define DMODEL  768
#define DINNER  1536
#define DSTATE  16
#define DTRANK  48
#define ROWS    (BATCH * SEQ)       // 8192
#define NP_XDBL 96                  // 80 padded to 96 (3 x 32)
#define KP_DT   64                  // 48 padded to 64 (2 x 32)

// ---- WMMA fragment helpers (wave32 layouts per CDNA5 ISA 7.12.2) ----------

// A fragment: 16x32 bf16 from row-major [M x K], top-left (m0,k0).
// Lane l: row m0+(l&15); elems 0..7 -> K=kb+0..7, elems 8..15 -> K=kb+16..23,
// kb = (l>=16) ? 8 : 0.  Two 16B contiguous loads.
__device__ __forceinline__ v16bf load_frag_a(const bf16* __restrict__ A, int lda,
                                             int m0, int k0, int lane) {
    const bf16* p = A + (size_t)(m0 + (lane & 15)) * lda + k0 + ((lane >> 4) << 3);
    v8bf lo = *(const v8bf*)p;
    v8bf hi = *(const v8bf*)(p + 16);
    v16bf f;
#pragma unroll
    for (int i = 0; i < 8; ++i) { f[i] = lo[i]; f[i + 8] = hi[i]; }
    return f;
}

// B fragment from the LDS-resident weight strip (32 rows x K cols, row-major,
// ld = K).  B = Wstrip^T.  Lane l: column nloc+(l&15); elems 0..7 ->
// K = k0+kb+0..7, elems 8..15 -> K = k0+kb+16..23, kb = (l>=16) ? 8 : 0.
__device__ __forceinline__ v16bf load_frag_b_lds(const bf16* __restrict__ bs, int K,
                                                 int nloc, int k0, int lane) {
    const bf16* p = bs + (size_t)(nloc + (lane & 15)) * K + k0 + ((lane >> 4) << 3);
    v8bf lo = *(const v8bf*)p;
    v8bf hi = *(const v8bf*)(p + 16);
    v16bf f;
#pragma unroll
    for (int i = 0; i < 8; ++i) { f[i] = lo[i]; f[i + 8] = hi[i]; }
    return f;
}

__device__ __forceinline__ v8f wmma_bf16(v16bf a, v16bf b, v8f c) {
    // (neg_a, A, neg_b, B, c_mod, C, reuse_a, reuse_b)
    return __builtin_amdgcn_wmma_f32_16x16x32_bf16(false, a, false, b,
                                                   (short)0, c, false, false);
}

// C/D 16x16 f32: lane l -> col n0+(l&15), vgpr r -> row m0 + (l>=16 ? 8:0) + r.
__device__ __forceinline__ void store_frag_c(float* __restrict__ C, int ldc,
                                             int m0, int n0, int lane, v8f c) {
    int n  = n0 + (lane & 15);
    int mb = m0 + ((lane >> 4) << 3);
#pragma unroll
    for (int r = 0; r < 8; ++r) C[(size_t)(mb + r) * ldc + n] = c[r];
}

// ---- GEMM: C[MxN] = A[MxK] * W[NxK]^T, bf16 in, f32 out -------------------
// Block = 256 threads = 8 waves; wave w -> M-tile (blockIdx.y*8+w)*32,
// N-tile blockIdx.x*32.  Dims always exact multiples (padded buffers), so no
// bounds checks and barriers are convergent.  Dynamic LDS = 32*K*2 bytes.
__global__ __launch_bounds__(256)
void gemm_bf16_wmma(const bf16* __restrict__ A, const bf16* __restrict__ W,
                    float* __restrict__ C, int M, int N, int K,
                    int lda, int ldb, int ldc) {
    extern __shared__ bf16 bsm[];            // 32 x K weight strip for this block
    const int lane = threadIdx.x & 31;
    const int wave = threadIdx.x >> 5;
    const int m0 = (blockIdx.y * 8 + wave) * 32;
    const int n0 = blockIdx.x * 32;

#if __has_builtin(__builtin_amdgcn_tensor_load_to_lds)
    // --- TDM: one 2-D descriptor DMAs the whole 32xK bf16 strip to LDS ---
    if (wave == 0) {
        uint64_t ga = (uint64_t)(uintptr_t)(W + (size_t)n0 * ldb); // tile start
        uint32_t la = (uint32_t)(uintptr_t)bsm;                    // LDS byte addr
        v4u g0;
        g0[0] = 1u;                                    // count=1 (valid D#)
        g0[1] = la;                                    // lds_addr
        g0[2] = (uint32_t)ga;                          // global_addr[31:0]
        g0[3] = (uint32_t)((ga >> 32) & 0x1FFFFFFu)    // global_addr[56:32]
              | (2u << 30);                            // type=2 ("image")
        v8i g1;
        g1[0] = (int)(1u << 16);                       // data_size=1 -> 2 bytes
        g1[1] = (int)(((uint32_t)K & 0xffffu) << 16);  // tensor_dim0[15:0]
        g1[2] = (int)((((uint32_t)K >> 16) & 0xffffu)  // tensor_dim0[31:16]
              | (32u << 16));                          // tensor_dim1 = 32 rows
        g1[3] = (int)(((uint32_t)K & 0xffffu) << 16);  // tile_dim0 = K
        g1[4] = 32;                                    // tile_dim1 = 32 (dim2=0)
        g1[5] = ldb;                                   // tensor_dim0_stride[31:0]
        g1[6] = 0;                                     // stride hi | dim1_stride lo
        g1[7] = 0;
        v4i z4 = {0, 0, 0, 0};                         // D# groups 2/3 (2-D tile)
        v8i z8 = {0, 0, 0, 0, 0, 0, 0, 0};
        __builtin_amdgcn_tensor_load_to_lds(g0, g1, z4, z4, z8, 0);
        __builtin_amdgcn_s_wait_tensorcnt(0);          // TENSORcnt == 0
    }
    __syncthreads();
#else
    // --- fallback: cooperative global->LDS copy (16B per thread per step) ---
    for (int e = threadIdx.x * 8; e < 32 * K; e += 256 * 8) {
        int r = e / K, c = e % K;                      // K multiple of 8
        *(v8bf*)(bsm + e) = *(const v8bf*)(W + (size_t)(n0 + r) * ldb + c);
    }
    __syncthreads();
#endif

    v8f c00 = {}, c01 = {}, c10 = {}, c11 = {};
    for (int k0 = 0; k0 < K; k0 += 32) {
        if (k0 + 32 < K) {                             // warm next A K-slab
            __builtin_prefetch(A + (size_t)(m0 + (lane & 15)) * lda + k0 + 32, 0, 1);
        }
        v16bf a0 = load_frag_a(A, lda, m0,      k0, lane);
        v16bf a1 = load_frag_a(A, lda, m0 + 16, k0, lane);
        v16bf b0 = load_frag_b_lds(bsm, K, 0,  k0, lane);
        v16bf b1 = load_frag_b_lds(bsm, K, 16, k0, lane);
        c00 = wmma_bf16(a0, b0, c00);
        c01 = wmma_bf16(a0, b1, c01);
        c10 = wmma_bf16(a1, b0, c10);
        c11 = wmma_bf16(a1, b1, c11);
    }
    store_frag_c(C, ldc, m0,      n0,      lane, c00);
    store_frag_c(C, ldc, m0,      n0 + 16, lane, c01);
    store_frag_c(C, ldc, m0 + 16, n0,      lane, c10);
    store_frag_c(C, ldc, m0 + 16, n0 + 16, lane, c11);
}

// ---- f32 -> bf16 with optional zero padding (rows/cols), src has own ld ---
__global__ void cvt_bf16_pad(const float* __restrict__ src, bf16* __restrict__ dst,
                             int dst_rows, int dst_cols,
                             int src_rows, int src_cols, int src_ld) {
    int idx = blockIdx.x * blockDim.x + threadIdx.x;
    if (idx >= dst_rows * dst_cols) return;
    int r = idx / dst_cols, c = idx % dst_cols;
    float v = (r < src_rows && c < src_cols) ? src[(size_t)r * src_ld + c] : 0.f;
    dst[idx] = (bf16)v;
}

// ---- depthwise causal conv1d (k=4, left pad 3) + SiLU ---------------------
__global__ void conv_silu(const float* __restrict__ xar, const float* __restrict__ cw,
                          const float* __restrict__ cb, float* __restrict__ hf,
                          bf16* __restrict__ hb) {
    int idx = blockIdx.x * blockDim.x + threadIdx.x;   // over ROWS*DINNER
    if (idx >= ROWS * DINNER) return;
    int d   = idx % DINNER;
    int row = idx / DINNER;
    int l   = row % SEQ;
    int b   = row / SEQ;
    float acc = cb[d];
#pragma unroll
    for (int j = 0; j < 4; ++j) {
        int ls = l - 3 + j;
        if (ls >= 0)
            acc += cw[d * 4 + j] * xar[(size_t)(b * SEQ + ls) * (2 * DINNER) + d];
    }
    float s = acc / (1.f + __expf(-acc));              // SiLU
    hf[idx] = s;
    hb[idx] = (bf16)s;
}

// ---- selective scan: one lane per (b,d) channel, 16 states in VGPRs -------
__global__ __launch_bounds__(128)
void selective_scan(const float* __restrict__ hf, const float* __restrict__ dtraw,
                    const float* __restrict__ bdt, const float* __restrict__ xdbl,
                    const float* __restrict__ alog, const float* __restrict__ Dp,
                    const float* __restrict__ xar, bf16* __restrict__ yb) {
    int tid = blockIdx.x * blockDim.x + threadIdx.x;
    if (tid >= BATCH * DINNER) return;
    int d = tid % DINNER;
    int b = tid / DINNER;

    float Arow[DSTATE];
#pragma unroll
    for (int n = 0; n < DSTATE; ++n) Arow[n] = -__expf(alog[d * DSTATE + n]);
    float st[DSTATE];
#pragma unroll
    for (int n = 0; n < DSTATE; ++n) st[n] = 0.f;
    const float Dd = Dp[d];
    const float bd = bdt[d];

    for (int t = 0; t < SEQ; ++t) {
        size_t row = (size_t)b * SEQ + t;
        float u  = hf[row * DINNER + d];
        float dr = dtraw[row * DINNER + d] + bd;
        float dt = (dr > 20.f) ? dr : log1pf(__expf(dr));   // softplus
        const float* bc = xdbl + row * NP_XDBL;             // dlt|B|C packed row
        float y = 0.f;
        float du = dt * u;
#pragma unroll
        for (int n = 0; n < DSTATE; ++n) {
            float dA = __expf(dt * Arow[n]);
            st[n] = dA * st[n] + du * bc[DTRANK + n];       // B_t cols 48..63
            y    += st[n] * bc[DTRANK + DSTATE + n];        // C_t cols 64..79
        }
        y += u * Dd;
        float r = xar[row * (2 * DINNER) + DINNER + d];     // res branch
        float g = r / (1.f + __expf(-r));                   // silu(res)
        yb[row * DINNER + d] = (bf16)(y * g);
    }
}

// ---------------------------------------------------------------------------
extern "C" void kernel_launch(void* const* d_in, const int* in_sizes, int n_in,
                              void* d_out, int out_size, void* d_ws, size_t ws_size,
                              hipStream_t stream) {
    (void)in_sizes; (void)n_in; (void)out_size; (void)ws_size;
    const float* x      = (const float*)d_in[0];   // (4,2048,768)
    const float* W_in   = (const float*)d_in[1];   // (3072,768)
    const float* conv_w = (const float*)d_in[2];   // (1536,1,4)
    const float* conv_b = (const float*)d_in[3];   // (1536,)
    const float* W_x    = (const float*)d_in[4];   // (80,1536)
    const float* W_dt   = (const float*)d_in[5];   // (1536,48)
    const float* b_dt   = (const float*)d_in[6];   // (1536,)
    const float* A_log  = (const float*)d_in[7];   // (1536,16)
    const float* Dvec   = (const float*)d_in[8];   // (1536,)
    const float* W_out  = (const float*)d_in[9];   // (768,1536)
    float* out = (float*)d_out;                    // (4,2048,768) f32

    // ---- workspace bump allocator (256B aligned) ----
    char* ws = (char*)d_ws;
    size_t off = 0;
    auto alloc = [&](size_t bytes) -> void* {
        void* p = ws + off;
        off += (bytes + 255) & ~(size_t)255;
        return p;
    };
    bf16*  xbf    = (bf16*) alloc((size_t)ROWS * DMODEL * 2);        // x in bf16
    bf16*  WinBf  = (bf16*) alloc((size_t)2 * DINNER * DMODEL * 2);  // (3072,768)
    bf16*  WxBf   = (bf16*) alloc((size_t)NP_XDBL * DINNER * 2);     // (96,1536) padded
    bf16*  WdtBf  = (bf16*) alloc((size_t)DINNER * KP_DT * 2);       // (1536,64) padded
    bf16*  WoutBf = (bf16*) alloc((size_t)DMODEL * DINNER * 2);      // (768,1536)
    float* xar    = (float*)alloc((size_t)ROWS * 2 * DINNER * 4);    // x_and_res
    float* hf     = (float*)alloc((size_t)ROWS * DINNER * 4);        // conv+silu f32
    bf16*  hb     = (bf16*) alloc((size_t)ROWS * DINNER * 2);        // conv+silu bf16
    float* xdbl   = (float*)alloc((size_t)ROWS * NP_XDBL * 4);       // dlt|B|C (pad 96)
    bf16*  dltb   = (bf16*) alloc((size_t)ROWS * KP_DT * 2);         // dlt bf16 (pad 64)
    float* dtraw  = (float*)alloc((size_t)ROWS * DINNER * 4);        // pre-softplus dt
    bf16*  yb     = (bf16*) alloc((size_t)ROWS * DINNER * 2);        // gated y bf16

    const int CVT = 256;
    auto cvt_n = [](int n) { return dim3((n + 255) / 256); };
    auto lds_bytes = [](int K) { return (size_t)32 * K * sizeof(bf16); };

    // ---- stage 0: f32 -> bf16 conversions (with padding) ----
    cvt_bf16_pad<<<cvt_n(ROWS * DMODEL), CVT, 0, stream>>>(x, xbf, ROWS, DMODEL, ROWS, DMODEL, DMODEL);
    cvt_bf16_pad<<<cvt_n(2 * DINNER * DMODEL), CVT, 0, stream>>>(W_in, WinBf, 2 * DINNER, DMODEL, 2 * DINNER, DMODEL, DMODEL);
    cvt_bf16_pad<<<cvt_n(NP_XDBL * DINNER), CVT, 0, stream>>>(W_x, WxBf, NP_XDBL, DINNER, DTRANK + 2 * DSTATE, DINNER, DINNER);
    cvt_bf16_pad<<<cvt_n(DINNER * KP_DT), CVT, 0, stream>>>(W_dt, WdtBf, DINNER, KP_DT, DINNER, DTRANK, DTRANK);
    cvt_bf16_pad<<<cvt_n(DMODEL * DINNER), CVT, 0, stream>>>(W_out, WoutBf, DMODEL, DINNER, DMODEL, DINNER, DINNER);

    // ---- stage 1: in_proj  x_and_res = x @ W_in^T  (8192x3072x768) ----
    gemm_bf16_wmma<<<dim3((2 * DINNER) / 32, ROWS / 256), 256, lds_bytes(DMODEL), stream>>>(
        xbf, WinBf, xar, ROWS, 2 * DINNER, DMODEL, DMODEL, DMODEL, 2 * DINNER);

    // ---- stage 2: depthwise causal conv1d + SiLU ----
    conv_silu<<<cvt_n(ROWS * DINNER), CVT, 0, stream>>>(xar, conv_w, conv_b, hf, hb);

    // ---- stage 3: x_proj  x_dbl = h @ W_x^T  (8192x96x1536, N padded) ----
    gemm_bf16_wmma<<<dim3(NP_XDBL / 32, ROWS / 256), 256, lds_bytes(DINNER), stream>>>(
        hb, WxBf, xdbl, ROWS, NP_XDBL, DINNER, DINNER, DINNER, NP_XDBL);

    // ---- stage 4: pack dlt (cols 0..47 of x_dbl) into bf16, K padded to 64 ----
    cvt_bf16_pad<<<cvt_n(ROWS * KP_DT), CVT, 0, stream>>>(xdbl, dltb, ROWS, KP_DT, ROWS, DTRANK, NP_XDBL);

    // ---- stage 5: dt_proj  dt_raw = dlt @ W_dt^T  (8192x1536x64) ----
    gemm_bf16_wmma<<<dim3(DINNER / 32, ROWS / 256), 256, lds_bytes(KP_DT), stream>>>(
        dltb, WdtBf, dtraw, ROWS, DINNER, KP_DT, KP_DT, KP_DT, DINNER);

    // ---- stage 6: fused softplus + selective scan + D residual + gating ----
    selective_scan<<<dim3((BATCH * DINNER) / 128), 128, 0, stream>>>(
        hf, dtraw, b_dt, xdbl, A_log, Dvec, xar, yb);

    // ---- stage 7: out_proj  out = y @ W_out^T  (8192x768x1536) ----
    gemm_bf16_wmma<<<dim3(DMODEL / 32, ROWS / 256), 256, lds_bytes(DINNER), stream>>>(
        yb, WoutBf, out, ROWS, DMODEL, DINNER, DINNER, DINNER, DMODEL);
}
